// COCA_13958643712727
// MI455X (gfx1250) — matrix operand order repaired and checked by reading, not compiled, not run on hardware
//
#include <hip/hip_runtime.h>
#include <math.h>

// ---------------------------------------------------------------------------
// MI455X (gfx1250) implementation. wave32, WMMA f32<-f16 16x16x32 matrix ops.
// Problem: B=2, NW=4, T=256, HID=256, OUT_CL=16, GN groups=4.  K is always 256
// for every matrix contraction, so the WMMA K-loops are fully unrolled
// (8 x v_wmma per kernel) and fragment loads use 128-bit global loads.
// ---------------------------------------------------------------------------

typedef __attribute__((ext_vector_type(16))) _Float16 v16h;
typedef __attribute__((ext_vector_type(8)))  float    v8f;

#define B_     2
#define NW_    4
#define T_     256
#define D_     256
#define OC_    16
#define GN_G   4
#define GC     64          // channels per group
#define HEADS  8           // B*NW
#define MROWS  2048        // B*NW*T
#define GEO    6

static __device__ __constant__ float GN_EPS_  = 1e-3f;
static __device__ __constant__ float LRELU    = 0.01f;
static __device__ __constant__ float SCL      = 0.0625f;              // 1/sqrt(256)
static __device__ __constant__ float TWO_PI_  = 6.283185307179586f;

// ---------------- WMMA fragment helpers (contiguous k order, k = 16*half + s) ----------------

// A fragment: 16x32 f16 from row-major f32 matrix (lda), tile at (row0, k0).
// lanes 0-15: row M=lane,    K in {0..15}
// lanes16-31: row M=lane-16, K in {16..31}
// Base (A + r*lda + k0 + half*16) is 16B aligned for all call sites.
__device__ __forceinline__ v16h load_a_f32(const float* __restrict__ A, int lda,
                                           int row0, int k0, int lane) {
  const int half = lane >> 4;
  const int r    = row0 + (lane & 15);
  const float4* p4 = reinterpret_cast<const float4*>(A + r * lda + k0 + half * 16);
  float4 x0 = p4[0], x1 = p4[1];   // K offsets 0..7   (within half)
  float4 y0 = p4[2], y1 = p4[3];   // K offsets 8..15  (within half)
  v16h a;
  a[0]=(_Float16)x0.x; a[1]=(_Float16)x0.y; a[2]=(_Float16)x0.z; a[3]=(_Float16)x0.w;
  a[4]=(_Float16)x1.x; a[5]=(_Float16)x1.y; a[6]=(_Float16)x1.z; a[7]=(_Float16)x1.w;
  a[8]=(_Float16)y0.x; a[9]=(_Float16)y0.y; a[10]=(_Float16)y0.z; a[11]=(_Float16)y0.w;
  a[12]=(_Float16)y1.x; a[13]=(_Float16)y1.y; a[14]=(_Float16)y1.z; a[15]=(_Float16)y1.w;
  return a;
}

// B fragment for A @ W^T:  B[k][n] = W[n][k], W row-major (N,K) with leading
// dim ldw.  lanes 0-15: col N=lane, K=k0..k0+15 ; lanes16-31: col N=lane-16,
// K=k0+16..k0+31.  Contiguous 16-float (64B, aligned) run per lane.
__device__ __forceinline__ v16h load_bT_f32(const float* __restrict__ W, int ldw,
                                            int n0, int k0, int lane) {
  const int c  = n0 + (lane & 15);
  const int kb = k0 + (lane >> 4) * 16;
  const float4* p4 = reinterpret_cast<const float4*>(W + c * ldw + kb);
  v16h b;
#pragma unroll
  for (int j = 0; j < 4; ++j) {
    float4 x = p4[j];
    b[4*j+0]=(_Float16)x.x; b[4*j+1]=(_Float16)x.y;
    b[4*j+2]=(_Float16)x.z; b[4*j+3]=(_Float16)x.w;
  }
  return b;
}

// block reduction over 256 threads, result broadcast to all threads
__device__ __forceinline__ float blk_sum(float v, float* red) {
  const int t = threadIdx.x;
  red[t] = v; __syncthreads();
  for (int o = 128; o > 0; o >>= 1) {
    if (t < o) red[t] += red[t + o];
    __syncthreads();
  }
  float r = red[0]; __syncthreads();
  return r;
}

// ---------------------------- group norm ------------------------------------

// one block per (b, group): mean/var over NW*T rows x 64 channels
__global__ __launch_bounds__(256)
void gn_stats(const float* __restrict__ X, float* __restrict__ stats) {
  const int b = blockIdx.x / GN_G;
  const int g = blockIdx.x % GN_G;
  const float* xb = X + b * NW_ * T_ * D_;
  const int n = NW_ * T_ * GC;
  float s = 0.f, ss = 0.f;
  for (int i = threadIdx.x; i < n; i += 256) {
    int row = i / GC, c = g * GC + (i % GC);
    float v = xb[row * D_ + c];
    s += v; ss += v * v;
  }
  __shared__ float red[256];
  float S  = blk_sum(s,  red);
  float SS = blk_sum(ss, red);
  if (threadIdx.x == 0) {
    float m   = S / (float)n;
    float var = SS / (float)n - m * m;
    stats[blockIdx.x * 2 + 0] = m;
    stats[blockIdx.x * 2 + 1] = rsqrtf(var + GN_EPS_);
  }
}

__global__ __launch_bounds__(256)
void gn_apply(const float* __restrict__ X, const float* __restrict__ stats,
              float* __restrict__ Y, int total) {
  int i = blockIdx.x * 256 + threadIdx.x;
  if (i >= total) return;
  int c = i % D_;
  int b = i / (NW_ * T_ * D_);
  int g = c / GC;
  float m = stats[(b * GN_G + g) * 2 + 0];
  float r = stats[(b * GN_G + g) * 2 + 1];
  Y[i] = (X[i] - m) * r;
}

// ------------------------- generic WMMA GEMM --------------------------------
// C[M,N] = act(A[M,K=256] @ W[N,256]^T + bias) + residual
// one 16x16 tile per wave; K-loop fully unrolled -> 8 back-to-back v_wmma.
template <bool ACT, bool BIAS, bool RES>
__global__ __launch_bounds__(256)
void gemm_wmma(const float* __restrict__ A, const float* __restrict__ W,
               const float* __restrict__ bias, const float* __restrict__ residual,
               float* __restrict__ C, int N) {
  const int lane = threadIdx.x & 31;
  const int wave = threadIdx.x >> 5;
  const int tile = blockIdx.x * 8 + wave;
  const int ntn  = N >> 4;
  const int m0   = (tile / ntn) << 4;
  const int n0   = (tile % ntn) << 4;
  v8f acc = {};
#pragma unroll
  for (int k0 = 0; k0 < D_; k0 += 32) {
    v16h a = load_a_f32(A, D_, m0, k0, lane);
    v16h b = load_bT_f32(W, D_, n0, k0, lane);
    acc = __builtin_amdgcn_wmma_f32_16x16x32_f16(false, a, false, b,
                                                 (short)0, acc, false, false);
  }
  const int c  = n0 + (lane & 15);
  const int r0 = m0 + ((lane >> 4) << 3);
  const float bb = BIAS ? bias[c] : 0.0f;
#pragma unroll
  for (int v = 0; v < 8; ++v) {
    float x = acc[v] + bb;
    if (ACT) x = (x >= 0.0f) ? x : LRELU * x;
    if (RES) x += residual[(r0 + v) * N + c];
    C[(r0 + v) * N + c] = x;
  }
}

// --------------- per-head Q.Q^T -> scaled squared distance ------------------
__global__ __launch_bounds__(256)
void qq_dist(const float* __restrict__ Q, const float* __restrict__ n2,
             float* __restrict__ Dist) {
  const int lane = threadIdx.x & 31;
  const int wave = threadIdx.x >> 5;
  const int tile = blockIdx.x * 8 + wave;     // 8 heads * 16 * 16 = 2048 tiles
  const int head = tile >> 8;
  const int rem  = tile & 255;
  const int m0   = (rem >> 4) << 4;
  const int n0   = (rem & 15) << 4;
  const float* Qh = Q + head * T_ * D_;
  v8f acc = {};
#pragma unroll
  for (int k0 = 0; k0 < D_; k0 += 32) {
    v16h a = load_a_f32(Qh, D_, m0, k0, lane);
    v16h b = load_bT_f32(Qh, D_, n0, k0, lane);   // B[k][n] = Q[n][k]
    acc = __builtin_amdgcn_wmma_f32_16x16x32_f16(false, a, false, b,
                                                 (short)0, acc, false, false);
  }
  const int c  = n0 + (lane & 15);
  const int r0 = m0 + ((lane >> 4) << 3);
  const float* n2h = n2 + head * T_;
  const float nc = n2h[c];
  float* Dh = Dist + head * T_ * T_;
#pragma unroll
  for (int v = 0; v < 8; ++v) {
    int r = r0 + v;
    Dh[r * T_ + c] = SCL * (n2h[r] + nc - 2.0f * acc[v]);
  }
}

__global__ __launch_bounds__(256)
void row_norm2(const float* __restrict__ Q, float* __restrict__ n2) {
  __shared__ float red[256];
  float v = Q[blockIdx.x * D_ + threadIdx.x];
  float s = blk_sum(v * v, red);
  if (threadIdx.x == 0) n2[blockIdx.x] = s;
}

// ----------- softmax(-d) + (af-min)/(max-min+eps), in place -----------------
__global__ __launch_bounds__(256)
void softmax_mm(float* __restrict__ A) {
  const int row = blockIdx.x;
  float d = A[row * T_ + threadIdx.x];
  __shared__ float r1[256], r2[256];
  r1[threadIdx.x] = d; r2[threadIdx.x] = d; __syncthreads();
  for (int o = 128; o > 0; o >>= 1) {
    if (threadIdx.x < o) {
      r1[threadIdx.x] = fminf(r1[threadIdx.x], r1[threadIdx.x + o]);
      r2[threadIdx.x] = fmaxf(r2[threadIdx.x], r2[threadIdx.x + o]);
    }
    __syncthreads();
  }
  float dmin = r1[0], dmax = r2[0];
  __syncthreads();
  float e = expf(-(d - dmin));
  float s = blk_sum(e, r1);
  float af    = e / s;
  float afmax = 1.0f / s;                    // element at d == dmin
  float afmin = expf(-(dmax - dmin)) / s;    // element at d == dmax
  A[row * T_ + threadIdx.x] = (af - afmin) / (afmax - afmin + 1e-8f);
}

// -------------------- compactness of af_m (per query row) -------------------
__global__ __launch_bounds__(256)
void comp_af(const float* __restrict__ AF, const float* __restrict__ GEOin,
             float* __restrict__ comp) {
  const int head = blockIdx.x >> 8;
  const int q    = blockIdx.x & 255;
  const int k    = threadIdx.x;
  const float* geo = GEOin + head * T_ * GEO;
  float mask = AF[(head * T_ + q) * T_ + k];
  mask = fminf(fmaxf(mask, 1e-5f), 0.99999f);
  const float dens = geo[k * GEO + 0], area = geo[k * GEO + 2];
  const float iner = geo[k * GEO + 3];
  const float cx = geo[k * GEO + 4], cy = geo[k * GEO + 5];
  const float qx = geo[q * GEO + 4], qy = geo[q * GEO + 5];
  const float dm = mask * dens, am = mask * area;
  const float mm = dm * am,     im = dm * iner;
  const float dx = qx - cx, dy = qy - cy;
  const float dist = dx * dx + dy * dy;
  __shared__ float sdm[256], sam[256], red[256];
  sdm[k] = dm; sam[k] = am; __syncthreads();
  float newiner = blk_sum(im + mm * dist, red);
  float ref1    = blk_sum(mm * am, red);
  float p = 0.f;
  for (int kk = 0; kk < k; ++kk) p += fminf(dm, sdm[kk]) * sam[kk];
  p *= 2.0f * am;
  float ref2 = blk_sum(p, red);
  if (k == 0) comp[blockIdx.x] = ((ref1 + ref2) / TWO_PI_) / newiner;
}

// ------------------- sequential clustering scan (15 steps) ------------------
__global__ __launch_bounds__(256)
void cluster_scan(const float* __restrict__ comp, const float* __restrict__ AF,
                  float* __restrict__ clm, float* __restrict__ clr,
                  float* __restrict__ scm) {
  const int head = blockIdx.x;
  const int k    = threadIdx.x;
  const float cq = comp[head * T_ + k];
  const float* afh = AF + head * T_ * T_;
  float log_sc = 0.f;
  __shared__ float sv[256];
  __shared__ int   si[256];
  for (int s = 0; s < OC_ - 1; ++s) {
    float score = cq * expf(log_sc);
    sv[k] = score; si[k] = k; __syncthreads();
    for (int o = 128; o > 0; o >>= 1) {
      if (k < o) {
        if (sv[k + o] > sv[k] || (sv[k + o] == sv[k] && si[k + o] < si[k])) {
          sv[k] = sv[k + o]; si[k] = si[k + o];
        }
      }
      __syncthreads();
    }
    const int idx = si[0];
    __syncthreads();
    float tm = afh[idx * T_ + k];
    tm = fminf(fmaxf(tm, 1e-5f), 0.99999f);
    clm[(head * OC_ + s) * T_ + k] = expf(log_sc) * tm; // exp(log_sc + log tm)
    log_sc += log1pf(-tm);
    if (k == 0) clr[head * (OC_ - 1) + s] = (float)idx;
  }
  const float sc = expf(log_sc);
  clm[(head * OC_ + OC_ - 1) * T_ + k] = sc;
  scm[head * T_ + k] = sc;
}

// --------------------- cluster aggregates (geo + msum) ----------------------
__global__ __launch_bounds__(256)
void cl_agg(const float* __restrict__ clm, const float* __restrict__ GEOin,
            float* __restrict__ clgeo, float* __restrict__ msum) {
  const int row  = blockIdx.x;          // head*16 + s
  const int head = row >> 4;
  const int k    = threadIdx.x;
  const float* geo = GEOin + head * T_ * GEO;
  const float m    = clm[row * T_ + k];
  const float dens = geo[k * GEO + 0], area = geo[k * GEO + 2];
  const float cx = geo[k * GEO + 4], cy = geo[k * GEO + 5];
  const float cd = m * dens, ca = m * area;
  const float cmass = cd * ca;
  __shared__ float red[256];
  float mass_s = blk_sum(cmass, red);
  float area_s = blk_sum(ca, red);
  float cxs    = blk_sum(cmass * cx, red);
  float cys    = blk_sum(cmass * cy, red);
  float ms     = blk_sum(m, red);
  if (k == 0) {
    float* g = clgeo + row * GEO;
    g[0] = mass_s / (area_s + 1e-8f);
    g[1] = mass_s;
    g[2] = area_s;
    g[4] = cxs / (mass_s + 1e-8f);
    g[5] = cys / (mass_s + 1e-8f);
    msum[row] = ms;
  }
}

// -------------------- compactness of cl_m (per cluster) ---------------------
__global__ __launch_bounds__(256)
void cl_comp_k(const float* __restrict__ clm, const float* __restrict__ GEOin,
               float* __restrict__ clgeo, float* __restrict__ clcomp) {
  const int row  = blockIdx.x;
  const int head = row >> 4;
  const int k    = threadIdx.x;
  const float* geo = GEOin + head * T_ * GEO;
  const float m    = clm[row * T_ + k];
  const float dens = geo[k * GEO + 0], area = geo[k * GEO + 2];
  const float iner = geo[k * GEO + 3];
  const float cx = geo[k * GEO + 4], cy = geo[k * GEO + 5];
  const float qx = clgeo[row * GEO + 4], qy = clgeo[row * GEO + 5];
  const float dm = m * dens, am = m * area;
  const float mm = dm * am,  im = dm * iner;
  const float dx = qx - cx, dy = qy - cy;
  const float dist = dx * dx + dy * dy;
  __shared__ float sdm[256], sam[256], red[256];
  sdm[k] = dm; sam[k] = am; __syncthreads();
  float newiner = blk_sum(im + mm * dist, red);
  float ref1    = blk_sum(mm * am, red);
  float p = 0.f;
  for (int kk = 0; kk < k; ++kk) p += fminf(dm, sdm[kk]) * sam[kk];
  p *= 2.0f * am;
  float ref2 = blk_sum(p, red);
  if (k == 0) {
    clgeo[row * GEO + 3] = newiner;                       // cl_iner
    clcomp[row] = ((ref1 + ref2) / TWO_PI_) / newiner;    // cl_comp
  }
}

// -------- per-head P(16xT) @ X(TxD) with P row-normalized (WMMA) ------------
template <bool TWO>
__global__ __launch_bounds__(256)
void ph_gemm(const float* __restrict__ CLM, const float* __restrict__ msum,
             const float* __restrict__ X1, const float* __restrict__ X2,
             float* __restrict__ Out) {
  const int lane = threadIdx.x & 31;
  const int wave = threadIdx.x >> 5;
  const int tile = blockIdx.x * 8 + wave;   // 8 heads * 16 n-tiles = 128
  const int head = tile >> 4;
  const int n0   = (tile & 15) << 4;
  const float* P  = CLM + head * OC_ * T_;
  const float* ms = msum + head * OC_;
  const float* x1 = X1 + head * T_ * D_;
  const float* x2 = X2 + head * T_ * D_;
  v8f acc = {};
  const int half = lane >> 4;
  const int rA   = lane & 15;
  const int c    = n0 + (lane & 15);
  const float rs = 1.0f / (ms[rA] + 1e-8f);     // cl_m_mean row scale
#pragma unroll
  for (int k0 = 0; k0 < T_; k0 += 32) {
    const float4* p4 = reinterpret_cast<const float4*>(P + rA * T_ + k0 + half * 16);
    float4 x0 = p4[0], x1f = p4[1], y0 = p4[2], y1 = p4[3];
    v16h a;
    a[0]=(_Float16)(x0.x*rs);  a[1]=(_Float16)(x0.y*rs);
    a[2]=(_Float16)(x0.z*rs);  a[3]=(_Float16)(x0.w*rs);
    a[4]=(_Float16)(x1f.x*rs); a[5]=(_Float16)(x1f.y*rs);
    a[6]=(_Float16)(x1f.z*rs); a[7]=(_Float16)(x1f.w*rs);
    a[8]=(_Float16)(y0.x*rs);  a[9]=(_Float16)(y0.y*rs);
    a[10]=(_Float16)(y0.z*rs); a[11]=(_Float16)(y0.w*rs);
    a[12]=(_Float16)(y1.x*rs); a[13]=(_Float16)(y1.y*rs);
    a[14]=(_Float16)(y1.z*rs); a[15]=(_Float16)(y1.w*rs);
    const int kb = k0 + (lane >> 4) * 16;
    v16h b;
#pragma unroll
    for (int i = 0; i < 16; ++i) {
      float xv = x1[(kb + i) * D_ + c];
      if (TWO) xv += x2[(kb + i) * D_ + c];
      b[i] = (_Float16)xv;
    }
    acc = __builtin_amdgcn_wmma_f32_16x16x32_f16(false, a, false, b,
                                                 (short)0, acc, false, false);
  }
  const int r0 = (lane >> 4) << 3;
  float* Oh = Out + head * OC_ * D_;
#pragma unroll
  for (int v = 0; v < 8; ++v) Oh[(r0 + v) * D_ + c] = acc[v];
}

// ---------------------------------------------------------------------------

extern "C" void kernel_launch(void* const* d_in, const int* in_sizes, int n_in,
                              void* d_out, int out_size, void* d_ws, size_t ws_size,
                              hipStream_t stream) {
  (void)in_sizes; (void)n_in; (void)out_size; (void)ws_size;
  const float* in_f   = (const float*)d_in[0];
  const float* in_geo = (const float*)d_in[1];
  const float* W1  = (const float*)d_in[2];
  const float* b1  = (const float*)d_in[3];
  const float* W2  = (const float*)d_in[4];
  const float* b2  = (const float*)d_in[5];
  const float* Wfc = (const float*)d_in[6];
  const float* bfc = (const float*)d_in[7];
  const float* Wq  = (const float*)d_in[8];
  const float* Wv  = (const float*)d_in[9];

  float* ws  = (float*)d_ws;
  float* out = (float*)d_out;

  // workspace layout (floats)
  float* stats0 = ws;                    // 16
  float* stats1 = ws + 16;               // 16
  float* f      = ws + 32;               // 2048*256
  float* h1     = f   + MROWS * D_;
  float* f_m    = h1  + MROWS * D_;
  float* f_v    = f_m + MROWS * D_;
  float* fqp    = f_v + MROWS * D_;
  float* n2     = fqp + MROWS * D_;      // 2048
  float* comp   = n2  + MROWS;           // 2048
  float* msum   = comp + MROWS;          // 128
  float* U      = msum + HEADS * OC_;    // 128*256

  // output layout (floats, reference return order)
  float* o_clf  = out;                          // (8,16,256)
  float* o_clm  = o_clf  + HEADS * OC_ * D_;    // (8,16,256)
  float* o_geo  = o_clm  + HEADS * OC_ * T_;    // (8,16,6)
  float* o_clr  = o_geo  + HEADS * OC_ * GEO;   // (8,15,1)
  float* o_comp = o_clr  + HEADS * (OC_ - 1);   // (8,16,1)
  float* o_af   = o_comp + HEADS * OC_;         // (8,256,256)
  float* o_scm  = o_af   + HEADS * T_ * T_;     // (8,1,256)
  float* o_tocl = o_scm  + HEADS * T_;          // (8,256,256)  f_q

  const int gemm_blocks  = (MROWS / 16) * (D_ / 16) / 8;          // 256
  const int gemm_blocksU = ((HEADS * OC_) / 16) * (D_ / 16) / 8;  // 16

  // 1) f = group_norm(in_f)
  gn_stats<<<B_ * GN_G, 256, 0, stream>>>(in_f, stats0);
  gn_apply<<<(MROWS * D_) / 256, 256, 0, stream>>>(in_f, stats0, f, MROWS * D_);
  // 2) MLP + projections (WMMA)
  gemm_wmma<true,  true,  false><<<gemm_blocks, 256, 0, stream>>>(f,  W1, b1, nullptr, h1,  D_);
  gemm_wmma<false, true,  false><<<gemm_blocks, 256, 0, stream>>>(h1, W2, b2, nullptr, f_m, D_);
  gemm_wmma<false, false, false><<<gemm_blocks, 256, 0, stream>>>(f,  Wv, nullptr, nullptr, f_v, D_);
  gemm_wmma<false, false, false><<<gemm_blocks, 256, 0, stream>>>(f,  Wq, nullptr, nullptr, fqp, D_);
  // 3) f_q = group_norm(f @ Wq^T)  -> to_cl output
  gn_stats<<<B_ * GN_G, 256, 0, stream>>>(fqp, stats1);
  gn_apply<<<(MROWS * D_) / 256, 256, 0, stream>>>(fqp, stats1, o_tocl, MROWS * D_);
  // 4) affinity: softmax(-scl * sq_dist(q, q)) + min/max normalize
  row_norm2<<<MROWS, 256, 0, stream>>>(o_tocl, n2);
  qq_dist<<<(HEADS * (T_ / 16) * (T_ / 16)) / 8, 256, 0, stream>>>(o_tocl, n2, o_af);
  softmax_mm<<<MROWS, 256, 0, stream>>>(o_af);
  // 5) compactness + clustering scan
  comp_af<<<MROWS, 256, 0, stream>>>(o_af, in_geo, comp);
  cluster_scan<<<HEADS, 256, 0, stream>>>(comp, o_af, o_clm, o_clr, o_scm);
  // 6) cluster aggregates / compactness
  cl_agg<<<HEADS * OC_, 256, 0, stream>>>(o_clm, in_geo, o_geo, msum);
  cl_comp_k<<<HEADS * OC_, 256, 0, stream>>>(o_clm, in_geo, o_geo, o_comp);
  // 7) cluster features:
  //    X1 = P @ (in_f + f_m)  -> cl_f ;  U = P @ f_v ;  cl_f += U @ Wfc^T + bfc
  ph_gemm<true ><<<(HEADS * 16) / 8, 256, 0, stream>>>(o_clm, msum, in_f, f_m, o_clf);
  ph_gemm<false><<<(HEADS * 16) / 8, 256, 0, stream>>>(o_clm, msum, f_v, in_f /*unused*/, U);
  gemm_wmma<false, true, true><<<gemm_blocksU, 256, 0, stream>>>(U, Wfc, bfc, o_clf, o_clf,
                                                                 D_);
}
